// TestFP8_GEMM_9268539424916
// MI455X (gfx1250) — 8-bit matrix operand order repaired and checked by reading, compiled, not run on hardware
//
#include <hip/hip_runtime.h>
#include <hip/hip_bf16.h>
#include <stdint.h>


typedef __attribute__((ext_vector_type(16))) int   v16i;
typedef __attribute__((ext_vector_type(8)))  float v8f;
typedef __attribute__((ext_vector_type(4)))  int   v4i;

typedef __attribute__((address_space(1))) v4i gv4;
typedef __attribute__((address_space(3))) v4i lv4;

#define MDIM 8192
#define KDIM 4096
#define NDIM 4096

#define BM 128
#define BN 128
#define BK 128
#define KTILES (KDIM / BK)

#if __has_builtin(__builtin_amdgcn_global_load_async_to_lds_b128)
#define USE_ASYNC_LDS 1
#else
#define USE_ASYNC_LDS 0
#endif

__device__ __forceinline__ gv4* to_global_v4(const void* p) {
    return (gv4*)(uintptr_t)p;
}
__device__ __forceinline__ lv4* to_lds_v4(void* p) {
    return (lv4*)(uint32_t)(uintptr_t)p;
}

__device__ __forceinline__ void wait_async_zero() {
#if __has_builtin(__builtin_amdgcn_s_wait_asynccnt)
    __builtin_amdgcn_s_wait_asynccnt(0);
#else
    asm volatile("s_wait_asynccnt 0x0" ::: "memory");
#endif
}


__device__ __forceinline__ uint32_t cvt_e4m3_sw(float x) {
    uint32_t u = __float_as_uint(x);
    uint32_t s = (u >> 24) & 0x80u;
    uint32_t a = u & 0x7FFFFFFFu;
    if (a > 0x7F800000u) return s | 0x7Fu;
    if (a >= 0x43E00000u) return s | 0x7Eu;
    int e = (int)(a >> 23) - 127;
    if (e < -9) return s;
    uint32_t m = (a & 0x7FFFFFu) | 0x800000u;
    int shift = 20 + ((e < -6) ? (-6 - e) : 0);
    uint32_t half = 1u << (shift - 1);
    uint32_t rnd  = half - 1u + ((m >> shift) & 1u);
    uint32_t q = (m + rnd) >> shift;
    int e8 = (e < -6) ? 0 : (e + 6);
    uint32_t v = ((uint32_t)e8 << 3) + q;
    if (v > 0x7Eu) v = 0x7Eu;
    return s | v;
}

__device__ __forceinline__ uint32_t pack4_e4m3(float4 f) {
#if __has_builtin(__builtin_amdgcn_cvt_pk_fp8_f32)
    int w = 0;
    w = __builtin_amdgcn_cvt_pk_fp8_f32(f.x, f.y, w, false);
    w = __builtin_amdgcn_cvt_pk_fp8_f32(f.z, f.w, w, true);
    return (uint32_t)w;
#else
    return cvt_e4m3_sw(f.x) | (cvt_e4m3_sw(f.y) << 8) |
           (cvt_e4m3_sw(f.z) << 16) | (cvt_e4m3_sw(f.w) << 24);
#endif
}

__global__ __launch_bounds__(256) void quant_fp8_e4m3(const float4* __restrict__ src,
                                                      uint4* __restrict__ dst, int n16) {
    int i = blockIdx.x * 256 + threadIdx.x;
    int stride = gridDim.x * 256;
    for (; i < n16; i += stride) {
        float4 f0 = src[i * 4 + 0];
        float4 f1 = src[i * 4 + 1];
        float4 f2 = src[i * 4 + 2];
        float4 f3 = src[i * 4 + 3];
        uint4 o;
        o.x = pack4_e4m3(f0);
        o.y = pack4_e4m3(f1);
        o.z = pack4_e4m3(f2);
        o.w = pack4_e4m3(f3);
        dst[i] = o;
    }
}


__global__ __launch_bounds__(256) void fp8_gemm_wmma(const uint8_t* __restrict__ A8,
                                                     const uint8_t* __restrict__ B8,
                                                     const __hip_bfloat16* __restrict__ bias,
                                                     float* __restrict__ out) {
    __shared__ uint4 As4[2][BM * BK / 16];
    __shared__ uint4 Bs4[2][BN * BK / 16];

    const int tid  = threadIdx.x;
    const int lane = tid & 31;
    const int wave = tid >> 5;
    const int wm   = wave & 3;
    const int wn   = wave >> 2;

    const int rowBase = blockIdx.x * BM;
    const int colBase = blockIdx.y * BN;

    const int lane15   = lane & 15;
    const int laneHalf = lane >> 4;

    const int fr = tid >> 3;
    const int fc = tid & 7;

    const v8f vzero = {0.f, 0.f, 0.f, 0.f, 0.f, 0.f, 0.f, 0.f};
    v8f acc[2][4];
#pragma unroll
    for (int mt = 0; mt < 2; ++mt)
#pragma unroll
        for (int nt = 0; nt < 4; ++nt) acc[mt][nt] = vzero;

#if USE_ASYNC_LDS
#define FILL_TILE(buf, kt)                                                                \
    {                                                                                     \
        const size_t ko = (size_t)(kt) * BK;                                              \
        _Pragma("unroll") for (int i = 0; i < 4; ++i) {                                   \
            const int r = fr + 32 * i;                                                    \
            __builtin_amdgcn_global_load_async_to_lds_b128(                               \
                to_global_v4(A8 + (size_t)(rowBase + r) * KDIM + ko + fc * 16),           \
                to_lds_v4(&As4[buf][r * 8 + fc]), 0, 0);                                  \
            __builtin_amdgcn_global_load_async_to_lds_b128(                               \
                to_global_v4(B8 + (size_t)(colBase + r) * KDIM + ko + fc * 16),           \
                to_lds_v4(&Bs4[buf][r * 8 + fc]), 0, 0);                                  \
        }                                                                                 \
    }
#define DRAIN_TILE(buf) wait_async_zero();
#else
    uint4 stageA[4], stageB[4];
#define FILL_TILE(buf, kt)                                                                \
    {                                                                                     \
        const size_t ko = (size_t)(kt) * BK;                                              \
        _Pragma("unroll") for (int i = 0; i < 4; ++i) {                                   \
            const int r = fr + 32 * i;                                                    \
            stageA[i] = *(const uint4*)(A8 + (size_t)(rowBase + r) * KDIM + ko + fc * 16);\
            stageB[i] = *(const uint4*)(B8 + (size_t)(colBase + r) * KDIM + ko + fc * 16);\
        }                                                                                 \
        _Pragma("unroll") for (int i = 0; i < 4; ++i) {                                   \
            As4[buf][(fr + 32 * i) * 8 + fc] = stageA[i];                                 \
            Bs4[buf][(fr + 32 * i) * 8 + fc] = stageB[i];                                 \
        }                                                                                 \
    }
#define DRAIN_TILE(buf)
#endif

    FILL_TILE(0, 0)
    DRAIN_TILE(0)
    __syncthreads();

    for (int kt = 0; kt < KTILES; ++kt) {
        const int buf = kt & 1;

        if (kt + 1 < KTILES) FILL_TILE(buf ^ 1, kt + 1)

        const int2* Alds = (const int2*)As4[buf];
        const int4* Blds = (const int4*)Bs4[buf];

        v16i afrag[2], bfrag[4];
#pragma unroll
        for (int mt = 0; mt < 2; ++mt) {
            const int row = wm * 32 + mt * 16 + lane15;
#pragma unroll
            for (int j = 0; j < 8; ++j) {
                const int k = (j >> 1) * 32 + laneHalf * 16 + (j & 1) * 8;
                int2 d = Alds[(row * BK + k) >> 3];
                afrag[mt][2 * j]     = d.x;
                afrag[mt][2 * j + 1] = d.y;
            }
        }
#pragma unroll
        for (int nt = 0; nt < 4; ++nt) {
            const int col = wn * 64 + nt * 16 + lane15;
#pragma unroll
            for (int g = 0; g < 4; ++g) {
                const int k = g * 32 + laneHalf * 16;
                int4 d = Blds[(col * BK + k) >> 4];
                bfrag[nt][4 * g]     = d.x;
                bfrag[nt][4 * g + 1] = d.y;
                bfrag[nt][4 * g + 2] = d.z;
                bfrag[nt][4 * g + 3] = d.w;
            }
        }

#pragma unroll
        for (int mt = 0; mt < 2; ++mt)
#pragma unroll
            for (int nt = 0; nt < 4; ++nt)
                acc[mt][nt] = __builtin_amdgcn_wmma_f32_16x16x128_fp8_fp8(
                    afrag[mt], bfrag[nt], (short)0, acc[mt][nt],
                     false,  false);

        if (kt + 1 < KTILES) DRAIN_TILE(buf ^ 1)
        __syncthreads();
    }

#pragma unroll
    for (int nt = 0; nt < 4; ++nt) {
        const int col = colBase + wn * 64 + nt * 16 + lane15;
        const float b = __bfloat162float(bias[col]);
#pragma unroll
        for (int mt = 0; mt < 2; ++mt) {
            const int row0 = rowBase + wm * 32 + mt * 16 + laneHalf * 8;
#pragma unroll
            for (int v = 0; v < 8; ++v)
                out[(size_t)(row0 + v) * NDIM + col] = acc[mt][nt][v] + b;
        }
    }
}


extern "C" void kernel_launch(void* const* d_in, const int* in_sizes, int n_in,
                              void* d_out, int out_size, void* d_ws, size_t ws_size,
                              hipStream_t stream) {
    (void)in_sizes; (void)n_in; (void)out_size; (void)ws_size;

    const float* inp           = (const float*)d_in[0];
    const float* weight        = (const float*)d_in[1];
    const __hip_bfloat16* bias = (const __hip_bfloat16*)d_in[2];
    float* out                 = (float*)d_out;

    uint8_t* a8 = (uint8_t*)d_ws;
    uint8_t* b8 = a8 + (size_t)MDIM * KDIM;

    const int n16a = MDIM * KDIM / 16;
    const int n16b = NDIM * KDIM / 16;
    quant_fp8_e4m3<<<4096, 256, 0, stream>>>((const float4*)inp,    (uint4*)a8, n16a);
    quant_fp8_e4m3<<<4096, 256, 0, stream>>>((const float4*)weight, (uint4*)b8, n16b);

    dim3 grid(MDIM / BM, NDIM / BN);
    fp8_gemm_wmma<<<grid, 256, 0, stream>>>(a8, b8, bias, out);
}
